// TMK_10067403342211
// MI455X (gfx1250) — matrix operand order repaired and checked by reading, not compiled, not run on hardware
//
#include <hip/hip_runtime.h>
#include <hip/hip_bf16.h>

typedef __attribute__((ext_vector_type(16))) _Float16 v16h;
typedef __attribute__((ext_vector_type(8)))  float    v8f;

#define N_ROWS   262144
#define M        512
#define D        10
#define BM       128         // rows per workgroup (8 waves x 16-row M-tiles)
#define NKB      16          // K blocks of 32 (K total = 512)
#define NT_ALL   32          // n-tiles total (512 / 16)
#define NT_CHUNK 4           // n-tiles per LDS-staged chunk
#define NCHUNK   (NT_ALL / NT_CHUNK)          // 8 chunks
#define CHUNK_HALVES (NT_CHUNK * NKB * 512)   // 32768 halves = 64 KB

// ---------------------------------------------------------------------------
// Pre-pass: convert chol_inv (f32 [512,512] row-major, element (k,n)) into
// f16 packed B-fragment blocks in workspace.
// Block (n_tile, kb) = 32 lanes x 16 halves contiguous (1024 B), n_tile-major.
// B 32x16 f16 layout (wave32): lane = nn + 16*((kk>>3)&1), half = (kk&7) + 8*((kk>>4)&1).
// ---------------------------------------------------------------------------
__global__ __launch_bounds__(256) void pack_b_kernel(const float* __restrict__ C,
                                                     _Float16* __restrict__ wsB) {
    int idx = blockIdx.x * 256 + threadIdx.x;     // 0 .. 512*512-1 == k*512+n
    int k = idx >> 9;
    int n = idx & (M - 1);
    int n_tile = n >> 4, nn = n & 15;
    int kb = k >> 5, kk = k & 31;
    int lane = nn + ((kk >> 3) & 1) * 16;
    int half = (kk & 7) + (((kk >> 4) & 1) << 3);
    int off  = ((n_tile * NKB + kb) << 9) + lane * 16 + half;
    wsB[off] = (_Float16)C[idx];
}

// issue per-thread share of a 64 KB global->LDS async copy (16 x b128)
__device__ __forceinline__ void issue_chunk_dma(const _Float16* gsrc,
                                                unsigned lds_base, int tid) {
    const char* g = (const char*)gsrc;
    #pragma unroll
    for (int i = 0; i < 16; ++i) {
        const unsigned off = (unsigned)(tid * 16 + i * 4096);
        unsigned long long ga = (unsigned long long)(g + off);
        unsigned la = lds_base + off;
        asm volatile("global_load_async_to_lds_b128 %0, %1, off"
                     :: "v"(la), "v"(ga) : "memory");
    }
}

// ---------------------------------------------------------------------------
// Main kernel.
//  phase 1: A = exp(-L1dist(x, pts)) as f16, stored A-fragment-swizzled in LDS
//           (overlapped with async DMA of the first B chunk).
//           A 16x32 f16 layout: lane = rr + 16*((kk>>3)&1),
//                               half = (kk&7) + 8*((kk>>4)&1).
//  phase 2: double-buffered: async-DMA chunk c+1 into LDS while doing the
//           WMMA GEMM on chunk c; A fragments (K=512) live in registers.
// ---------------------------------------------------------------------------
__global__ __launch_bounds__(256) void laplace_wmma_kernel(
    const float* __restrict__ x,        // [N_ROWS, 10]
    const float* __restrict__ pts,      // [512, 10]
    const _Float16* __restrict__ wsB,   // packed B fragments (512 KB)
    float* __restrict__ out)            // [N_ROWS, 512]
{
    __shared__ __align__(32) _Float16 sA[BM * M];              // 128 KB
    __shared__ __align__(32) _Float16 sB[2][CHUNK_HALVES];     // 2 x 64 KB
    __shared__ float sP[M * D];                                // 20 KB pts

    const int tid  = threadIdx.x;
    const int lane = tid & 31;
    const int wid  = tid >> 5;          // 0..7 == M-tile of this wave
    const long long row_base = (long long)blockIdx.x * BM;

    const unsigned sB_lds0 = (unsigned)(size_t)&sB[0][0];
    const unsigned sB_lds1 = (unsigned)(size_t)&sB[1][0];

    // kick off DMA of B chunk 0 -> buffer 0; overlaps all of phase 1
    issue_chunk_dma(wsB, sB_lds0, tid);

    // stage pts_set into LDS
    for (int i = tid; i < M * D; i += 256) sP[i] = pts[i];
    __syncthreads();

    // ---------------- phase 1: K_tile = exp(-sum |x - p|) --------------------
    {
        const int r  = tid >> 1;            // row in tile: 0..127 (2 thr/row)
        const int c0 = tid & 1;
        const float* xr = x + (row_base + r) * D;
        float xv[D];
        #pragma unroll
        for (int j = 0; j < D; ++j) xv[j] = xr[j];

        #pragma unroll 4
        for (int q = 0; q < M / 2; ++q) {
            const int c = c0 + q * 2;       // column 0..511
            const float* p = &sP[c * D];
            float dist = 0.0f;
            #pragma unroll
            for (int j = 0; j < D; ++j) dist += __builtin_fabsf(xv[j] - p[j]);
            const float v = __expf(-dist);  // LENGTHSCALE == 1
            const int kb = c >> 5, kk = c & 31;
            const int blk  = ((r >> 4) * NKB + kb);
            const int la   = (r & 15) + (((kk >> 3) & 1) << 4);
            const int half = (kk & 7) + (((kk >> 4) & 1) << 3);
            sA[(blk << 9) + la * 16 + half] = (_Float16)v;
        }
    }
    asm volatile("s_wait_asynccnt 0x0" ::: "memory");   // chunk 0 DMA done
    __syncthreads();                                    // sA + sB[0] visible

    // Hold this wave's 16 A fragments (full K=512) in registers (128 VGPRs).
    v16h afrag[NKB];
    {
        const _Float16* aBase = &sA[((wid * NKB) << 9) + lane * 16];
        #pragma unroll
        for (int kb = 0; kb < NKB; ++kb)
            afrag[kb] = *(const v16h*)(aBase + (kb << 9));
    }

    // ---------------- phase 2: double-buffered WMMA GEMM ---------------------
    #pragma unroll 1
    for (int chunk = 0; chunk < NCHUNK; ++chunk) {
        // prefetch next chunk into the other buffer while we compute
        if (chunk + 1 < NCHUNK) {
            issue_chunk_dma(wsB + (size_t)(chunk + 1) * CHUNK_HALVES,
                            ((chunk + 1) & 1) ? sB_lds1 : sB_lds0, tid);
        }

        const _Float16* bChunk = &sB[chunk & 1][0];
        #pragma unroll 1
        for (int nt = 0; nt < NT_CHUNK; ++nt) {
            const int n_tile = chunk * NT_CHUNK + nt;
            v8f acc = {};
            const _Float16* bBase = bChunk + ((nt * NKB) << 9) + lane * 16;
            #pragma unroll
            for (int kb = 0; kb < NKB; ++kb) {
                v16h b = *(const v16h*)(bBase + (kb << 9));
                acc = __builtin_amdgcn_wmma_f32_16x16x32_f16(
                    false, afrag[kb], false, b, (short)0, acc, false, false);
            }
            // D layout: VGPR g, lane L -> row = g + 8*(L>>4), col = L&15
            const int col = n_tile * 16 + (lane & 15);
            const long long row0 = row_base + wid * 16 + ((lane >> 4) << 3);
            float* o = out + row0 * M + col;
            #pragma unroll
            for (int g = 0; g < 8; ++g) o[(long long)g * M] = acc[g];
        }

        // next chunk's DMA must be complete, and all waves done reading
        asm volatile("s_wait_asynccnt 0x0" ::: "memory");
        __syncthreads();
    }
}

extern "C" void kernel_launch(void* const* d_in, const int* in_sizes, int n_in,
                              void* d_out, int out_size, void* d_ws, size_t ws_size,
                              hipStream_t stream) {
    const float* x        = (const float*)d_in[0];   // [262144,10]
    const float* pts_set  = (const float*)d_in[1];   // [512,10]
    const float* chol_inv = (const float*)d_in[2];   // [512,512]
    float* out            = (float*)d_out;           // [262144,512]
    _Float16* wsB         = (_Float16*)d_ws;         // 512 KB packed B

    pack_b_kernel<<<(M * M) / 256, 256, 0, stream>>>(chol_inv, wsB);
    laplace_wmma_kernel<<<N_ROWS / BM, 256, 0, stream>>>(x, pts_set, wsB, out);
}
